// CNNHTMModel_33775622816068
// MI455X (gfx1250) — matrix operand order repaired and checked by reading, not compiled, not run on hardware
//
#include <hip/hip_runtime.h>
#include <hip/hip_bf16.h>

// ---------------------------------------------------------------------------
// CDNA5 (gfx1250, wave32) implementation of the HTM attention model.
// CNN branch of the reference is dead code (result discarded) -> skipped.
// All large GEMMs: v_wmma_f32_16x16x32_bf16, bf16 operands pre-packed so each
// fragment is contiguous 16B vector loads (no per-iteration cvt / strided b32).
// ---------------------------------------------------------------------------

typedef __attribute__((ext_vector_type(16))) __bf16 v16bf;
typedef __attribute__((ext_vector_type(8)))  __bf16 v8bf;
typedef __attribute__((ext_vector_type(8)))  float  v8f;

union bf16x16 { v16bf v; v8bf h[2]; };

#define NEG_MAX (-3.402823466e38f)

static __device__ inline v16bf load_frag16(const __bf16* p0, const __bf16* p1) {
    bf16x16 u;
    u.h[0] = *(const v8bf*)p0;
    u.h[1] = *(const v8bf*)p1;
    return u.v;
}

// --------------------------------------------------------------------------
// Positional encoding table pos[c][d], c in [0,32), d in [0,512)
// --------------------------------------------------------------------------
__global__ void pos_kernel(float* __restrict__ posT) {
    int t = blockIdx.x * blockDim.x + threadIdx.x;   // 0..8191
    if (t >= 32 * 256) return;
    int c = t >> 8;
    int f = t & 255;
    float inv = __powf(10000.0f, -(float)(2 * f) / 512.0f);
    float ang = (float)(31 - c) * inv;
    posT[c * 512 + f]       = __sinf(ang);
    posT[c * 512 + 256 + f] = __cosf(ang);
}

// --------------------------------------------------------------------------
// f32 -> bf16 elementwise
// --------------------------------------------------------------------------
__global__ __launch_bounds__(256) void convert_bf16(
    const float* __restrict__ X, __bf16* __restrict__ Y, int n)
{
    int i = blockIdx.x * 256 + threadIdx.x;
    if (i < n) Y[i] = (__bf16)X[i];
}

// --------------------------------------------------------------------------
// Transpose + convert: Bt[n*K + k] = (bf16) B[k*N + n]   (B is K x N, f32)
// 32x32 tiles through LDS for coalescing on both sides.
// --------------------------------------------------------------------------
__global__ __launch_bounds__(256) void transpose_to_bf16(
    const float* __restrict__ B, __bf16* __restrict__ Bt, int K, int N)
{
    __shared__ float tile[32][33];
    int k0 = blockIdx.y * 32, n0 = blockIdx.x * 32;
    int tx = threadIdx.x & 31, ty = threadIdx.x >> 5;   // 32 x 8
    for (int i = ty; i < 32; i += 8) {
        int k = k0 + i, n = n0 + tx;
        tile[i][tx] = (k < K && n < N) ? B[(size_t)k * N + n] : 0.0f;
    }
    __syncthreads();
    for (int i = ty; i < 32; i += 8) {
        int n = n0 + i, k = k0 + tx;
        if (n < N && k < K) Bt[(size_t)n * K + k] = (__bf16)tile[tx][i];
    }
}

// --------------------------------------------------------------------------
// Masked mean over each chunk of 32 rows -> summarized (2500 x 512, bf16)
// --------------------------------------------------------------------------
__global__ __launch_bounds__(256) void summarize_kernel(
    const float* __restrict__ memories, const unsigned char* __restrict__ mask,
    __bf16* __restrict__ summB, int* __restrict__ cany)
{
    int blk = blockIdx.x;                 // b*625 + n
    int b = blk / 625, n = blk % 625;
    const float* base = memories + ((size_t)b * 20000 + (size_t)n * 32) * 512;
    const unsigned char* mrow = mask + (size_t)b * 20000 + (size_t)n * 32;
    int tid = threadIdx.x;

    int cnt = 0;
    unsigned int mbits = 0;
#pragma unroll
    for (int c = 0; c < 32; ++c)
        if (mrow[c]) { ++cnt; mbits |= (1u << c); }
    float inv = 1.0f / ((float)cnt + 1e-5f);

    for (int d = tid; d < 512; d += 256) {
        float s = 0.0f;
#pragma unroll
        for (int c = 0; c < 32; ++c)
            if (mbits & (1u << c)) s += base[(size_t)c * 512 + d];
        summB[(size_t)blk * 512 + d] = (__bf16)(s * inv);
    }
    if (tid == 0) cany[blk] = (cnt > 0) ? 1 : 0;
}

// --------------------------------------------------------------------------
// Unified bf16 GEMM:  C[MxN] = (A[MxK] @ Bt[NxK]^T) * scale + bias
// A, Bt bf16 row-major; one wave per 16x32 output tile (2 WMMAs / k-step,
// A fragment reused). OOB rows/cols are pointer-clamped (stores guarded),
// so the inner loop has no divergence. All fragment loads are 16B vectors.
// --------------------------------------------------------------------------
__global__ __launch_bounds__(32) void gemm_bf16(
    const __bf16* __restrict__ A, const __bf16* __restrict__ Bt,
    const float* __restrict__ bias, void* __restrict__ Cv,
    int M, int N, int K, float scale, int outBf16,
    size_t strideA, size_t strideB, size_t strideC)
{
    int lane = threadIdx.x & 31;
    int half = lane >> 4, l16 = lane & 15;
    int mrow = blockIdx.y * 16 + l16;
    int nc0  = blockIdx.x * 32 + l16;
    int nc1  = nc0 + 16;

    const __bf16* Ab = A  + (size_t)blockIdx.z * strideA;
    const __bf16* Bb = Bt + (size_t)blockIdx.z * strideB;

    const __bf16* Ar  = Ab + (size_t)(mrow < M ? mrow : 0) * K + half * 8;
    const __bf16* Br0 = Bb + (size_t)(nc0  < N ? nc0  : 0) * K + half * 8;
    const __bf16* Br1 = Bb + (size_t)(nc1  < N ? nc1  : 0) * K + half * 8;

    v8f acc0 = {0.f, 0.f, 0.f, 0.f, 0.f, 0.f, 0.f, 0.f};
    v8f acc1 = {0.f, 0.f, 0.f, 0.f, 0.f, 0.f, 0.f, 0.f};

    for (int kb = 0; kb < K; kb += 32) {
        v16bf a  = load_frag16(Ar  + kb, Ar  + kb + 16);
        v16bf b0 = load_frag16(Br0 + kb, Br0 + kb + 16);
        v16bf b1 = load_frag16(Br1 + kb, Br1 + kb + 16);
        acc0 = __builtin_amdgcn_wmma_f32_16x16x32_bf16(
            false, a, false, b0, (short)0, acc0, false, false);
        acc1 = __builtin_amdgcn_wmma_f32_16x16x32_bf16(
            false, a, false, b1, (short)0, acc1, false, false);
    }

    int rbase = blockIdx.y * 16 + half * 8;
    float bv0 = (bias && nc0 < N) ? bias[nc0] : 0.0f;
    float bv1 = (bias && nc1 < N) ? bias[nc1] : 0.0f;
    if (outBf16) {
        __bf16* Cb = (__bf16*)Cv + (size_t)blockIdx.z * strideC;
#pragma unroll
        for (int v = 0; v < 8; ++v) {
            int r = rbase + v;
            if (r < M) {
                if (nc0 < N) Cb[(size_t)r * N + nc0] = (__bf16)(acc0[v] * scale + bv0);
                if (nc1 < N) Cb[(size_t)r * N + nc1] = (__bf16)(acc1[v] * scale + bv1);
            }
        }
    } else {
        float* Cb = (float*)Cv + (size_t)blockIdx.z * strideC;
#pragma unroll
        for (int v = 0; v < 8; ++v) {
            int r = rbase + v;
            if (r < M) {
                if (nc0 < N) Cb[(size_t)r * N + nc0] = acc0[v] * scale + bv0;
                if (nc1 < N) Cb[(size_t)r * N + nc1] = acc1[v] * scale + bv1;
            }
        }
    }
}

// --------------------------------------------------------------------------
// Per (b,i) row: top-8 of 625 logits (lowest-index tie break, as lax.top_k),
// then softmax over the 8 picked logits -> weights.
// --------------------------------------------------------------------------
__global__ __launch_bounds__(128) void topk_kernel(
    const float* __restrict__ logits, const int* __restrict__ cany,
    int* __restrict__ topk, float* __restrict__ wts)
{
    int row = blockIdx.x;     // b*128 + i
    int b   = row >> 7;
    int tid = threadIdx.x;

    __shared__ float lg[640];
    __shared__ float rv[128];
    __shared__ int   ri[128];
    __shared__ float tv[8];

    for (int n = tid; n < 640; n += 128)
        lg[n] = (n < 625 && cany[b * 625 + n]) ? logits[(size_t)row * 625 + n]
                                               : NEG_MAX;
    __syncthreads();

    for (int t = 0; t < 8; ++t) {
        float best = NEG_MAX;
        int bi = 0x7fffffff;
        for (int n = tid; n < 625; n += 128) {
            float v = lg[n];
            if (v > best || (v == best && n < bi)) { best = v; bi = n; }
        }
        rv[tid] = best; ri[tid] = bi;
        __syncthreads();
        for (int s = 64; s > 0; s >>= 1) {
            if (tid < s) {
                float v2 = rv[tid + s]; int i2 = ri[tid + s];
                if (v2 > rv[tid] || (v2 == rv[tid] && i2 < ri[tid])) {
                    rv[tid] = v2; ri[tid] = i2;
                }
            }
            __syncthreads();
        }
        if (tid == 0) {
            topk[row * 8 + t] = ri[0];
            tv[t] = rv[0];
            lg[ri[0]] = NEG_MAX;
        }
        __syncthreads();
    }

    if (tid == 0) {
        float m = tv[0];
#pragma unroll
        for (int t = 1; t < 8; ++t) m = fmaxf(m, tv[t]);
        float e[8], s = 0.0f;
#pragma unroll
        for (int t = 0; t < 8; ++t) { e[t] = __expf(tv[t] - m); s += e[t]; }
#pragma unroll
        for (int t = 0; t < 8; ++t) wts[row * 8 + t] = e[t] / s;
    }
}

// --------------------------------------------------------------------------
// Fused attention kernel, one block per (b, i, k) selection (4096 blocks).
//  1. stage selected chunk + pos as bf16 in LDS (A: 32x512)
//  2. KV = A @ kv_w (32x1024) with WMMA; B frags contiguous from kvT bf16
//  3. per head: sim = q.K^T, masked softmax (wave32 shfl), o = attn.V (bf16)
// Dynamic LDS: 512f q + 256f attn + 32*512 bf16 A + 32*1024 bf16 KV = 99 KB
// --------------------------------------------------------------------------
__global__ __launch_bounds__(256) void attn_kernel(
    const float* __restrict__ memories, const unsigned char* __restrict__ mask,
    const __bf16* __restrict__ kvT,      // (1024 x 512) bf16, pre-transposed
    const float* __restrict__ qbuf, const float* __restrict__ posT,
    const int* __restrict__ topk, __bf16* __restrict__ obufB)
{
    extern __shared__ char smem[];
    float*  qLds    = (float*)smem;                 // 512
    float*  attnLds = qLds + 512;                   // 256 (8 heads x 32)
    __bf16* Alds    = (__bf16*)(attnLds + 256);     // 32 x 512
    __bf16* KVlds   = Alds + 32 * 512;              // 32 x 1024 (K | V)

    int b  = blockIdx.z;
    int qi = blockIdx.y;
    int ks = blockIdx.x;
    int row = (b * 128 + qi) * 8 + ks;
    int sel = topk[row];

    const float* chunk = memories + ((size_t)b * 20000 + (size_t)sel * 32) * 512;
    const unsigned char* cmask = mask + (size_t)b * 20000 + (size_t)sel * 32;
    int tid = threadIdx.x;

    for (int idx = tid; idx < 32 * 512; idx += 256)
        Alds[idx] = (__bf16)(chunk[idx] + posT[idx]);
    for (int idx = tid; idx < 512; idx += 256)
        qLds[idx] = qbuf[(size_t)(b * 128 + qi) * 512 + idx];
    __syncthreads();

    // ---- KV projection: (32x512) @ (512x1024) with WMMA ----
    int wave = tid >> 5, lane = tid & 31;
    int half = lane >> 4, l16 = lane & 15;

    const __bf16* A0 = Alds + (size_t)l16 * 512 + half * 8;
    const __bf16* A1 = Alds + (size_t)(16 + l16) * 512 + half * 8;

    for (int nt = wave; nt < 64; nt += 8) {
        int ncol = nt * 16 + l16;
        const __bf16* Br = kvT + (size_t)ncol * 512 + half * 8;
        v8f acc0 = {0.f, 0.f, 0.f, 0.f, 0.f, 0.f, 0.f, 0.f};
        v8f acc1 = {0.f, 0.f, 0.f, 0.f, 0.f, 0.f, 0.f, 0.f};
        for (int kb = 0; kb < 512; kb += 32) {
            v16bf bb = load_frag16(Br + kb, Br + kb + 16);
            v16bf a0 = load_frag16(A0 + kb, A0 + kb + 16);
            v16bf a1 = load_frag16(A1 + kb, A1 + kb + 16);
            acc0 = __builtin_amdgcn_wmma_f32_16x16x32_bf16(
                false, a0, false, bb, (short)0, acc0, false, false);
            acc1 = __builtin_amdgcn_wmma_f32_16x16x32_bf16(
                false, a1, false, bb, (short)0, acc1, false, false);
        }
#pragma unroll
        for (int v = 0; v < 8; ++v) {
            KVlds[(size_t)(half * 8 + v) * 1024 + ncol]      = (__bf16)acc0[v];
            KVlds[(size_t)(16 + half * 8 + v) * 1024 + ncol] = (__bf16)acc1[v];
        }
    }
    __syncthreads();

    // ---- attention: wave = head h, lane = chunk position c ----
    int h = wave, c = lane;
    float sim = 0.0f;
#pragma unroll
    for (int e = 0; e < 64; ++e)
        sim += qLds[h * 64 + e] * (float)KVlds[(size_t)c * 1024 + h * 64 + e];
    if (!cmask[c]) sim = NEG_MAX;

    float mx = sim;
#pragma unroll
    for (int s = 16; s > 0; s >>= 1) mx = fmaxf(mx, __shfl_xor(mx, s, 32));
    float p = __expf(sim - mx);
    float sum = p;
#pragma unroll
    for (int s = 16; s > 0; s >>= 1) sum += __shfl_xor(sum, s, 32);
    attnLds[h * 32 + c] = p / sum;
    __syncthreads();

    // ---- o = attn @ V : lane covers e = lane and lane+32 of this head ----
    float o0 = 0.0f, o1 = 0.0f;
#pragma unroll
    for (int cc = 0; cc < 32; ++cc) {
        float av = attnLds[h * 32 + cc];
        o0 += av * (float)KVlds[(size_t)cc * 1024 + 512 + h * 64 + lane];
        o1 += av * (float)KVlds[(size_t)cc * 1024 + 512 + h * 64 + 32 + lane];
    }
    obufB[(size_t)row * 512 + h * 64 + lane]      = (__bf16)o0;
    obufB[(size_t)row * 512 + h * 64 + 32 + lane] = (__bf16)o1;
}

// --------------------------------------------------------------------------
// htm = sum_k weights[k] * oproj[b,i,k,:]; out = htm @ fc2_w + fc2_b
// --------------------------------------------------------------------------
__global__ __launch_bounds__(256) void final_kernel(
    const float* __restrict__ oproj, const float* __restrict__ wts,
    const float* __restrict__ fc2_w, const float* __restrict__ fc2_b,
    float* __restrict__ out)
{
    int row = blockIdx.x;     // b*128 + i
    int tid = threadIdx.x;
    __shared__ float htm[512];

    const float* w8 = wts + (size_t)row * 8;
    for (int d = tid; d < 512; d += 256) {
        float s = 0.0f;
#pragma unroll
        for (int k = 0; k < 8; ++k)
            s += w8[k] * oproj[((size_t)row * 8 + k) * 512 + d];
        htm[d] = s;
    }
    __syncthreads();

    if (tid < 5) {
        float s = 0.0f;
        for (int d = 0; d < 512; ++d) s += htm[d] * fc2_w[(size_t)d * 5 + tid];
        out[(size_t)row * 5 + tid] = s + fc2_b[tid];
    }
}

// --------------------------------------------------------------------------
extern "C" void kernel_launch(void* const* d_in, const int* in_sizes, int n_in,
                              void* d_out, int out_size, void* d_ws, size_t ws_size,
                              hipStream_t stream)
{
    (void)in_sizes; (void)n_in; (void)out_size; (void)ws_size;

    // setup_inputs order; CNN inputs (0, 4..9) are dead code in the reference.
    const float*         queries  = (const float*)d_in[1];   // (4,128,512)
    const float*         memories = (const float*)d_in[2];   // (4,20000,512)
    const unsigned char* mask     = (const unsigned char*)d_in[3];
    const float*         sq_w  = (const float*)d_in[10];
    const float*         sq_b  = (const float*)d_in[11];
    const float*         sk_w  = (const float*)d_in[12];
    const float*         sk_b  = (const float*)d_in[13];
    const float*         q_w   = (const float*)d_in[14];
    const float*         kv_w  = (const float*)d_in[15];     // (512,1024)
    const float*         out_w = (const float*)d_in[16];     // (512,512)
    const float*         out_b = (const float*)d_in[17];
    const float*         fc2_w = (const float*)d_in[18];     // (512,5)
    const float*         fc2_b = (const float*)d_in[19];
    float* out = (float*)d_out;                              // (4,128,5)

    // ---- workspace carve-up (256B-aligned chunks) ----
    char* p = (char*)d_ws;
    auto carve = [&p](size_t bytes) {
        char* r = p;
        p += (bytes + 255) & ~(size_t)255;
        return r;
    };
    float*  posT  = (float*)carve(16384 * 4);                 // 32x512
    __bf16* summB = (__bf16*)carve((size_t)1280000 * 2);      // 2500x512
    int*    cany  = (int*)carve(2500 * 4);
    __bf16* qryB  = (__bf16*)carve((size_t)262144 * 2);       // 512x512
    __bf16* sqB   = (__bf16*)carve((size_t)262144 * 2);       // 512x512
    __bf16* skB   = (__bf16*)carve((size_t)1280000 * 2);      // 2500x512
    float*  qbuf  = (float*)carve((size_t)262144 * 4);        // 512x512
    float*  lgt   = (float*)carve((size_t)320000 * 4);        // 4x128x625
    int*    tki   = (int*)carve(4096 * 4);
    float*  wts   = (float*)carve(4096 * 4);
    __bf16* obufB = (__bf16*)carve((size_t)2097152 * 2);      // 4096x512
    float*  opro  = (float*)carve((size_t)2097152 * 4);       // 4096x512
    __bf16* sqwT  = (__bf16*)carve((size_t)262144 * 2);       // 512x512 ^T
    __bf16* skwT  = (__bf16*)carve((size_t)262144 * 2);
    __bf16* qwT   = (__bf16*)carve((size_t)262144 * 2);
    __bf16* outwT = (__bf16*)carve((size_t)262144 * 2);
    __bf16* kvwT  = (__bf16*)carve((size_t)524288 * 2);       // 1024x512 ^T

    // 1. positional table + operand packing (bf16, weights transposed to NxK)
    pos_kernel<<<32, 256, 0, stream>>>(posT);
    convert_bf16<<<1024, 256, 0, stream>>>(queries, qryB, 262144);
    transpose_to_bf16<<<dim3(16, 16), 256, 0, stream>>>(sq_w, sqwT, 512, 512);
    transpose_to_bf16<<<dim3(16, 16), 256, 0, stream>>>(sk_w, skwT, 512, 512);
    transpose_to_bf16<<<dim3(16, 16), 256, 0, stream>>>(q_w, qwT, 512, 512);
    transpose_to_bf16<<<dim3(16, 16), 256, 0, stream>>>(out_w, outwT, 512, 512);
    transpose_to_bf16<<<dim3(32, 16), 256, 0, stream>>>(kv_w, kvwT, 512, 1024);

    // 2. masked chunk means (one HBM pass over memories; warms L2)
    summarize_kernel<<<2500, 256, 0, stream>>>(memories, mask, summB, cany);

    // 3. projections: sq, sk (bf16 out, feed logits), q (f32 out, feeds attn)
    gemm_bf16<<<dim3(16, 32), 32, 0, stream>>>(
        qryB, sqwT, sq_b, sqB, 512, 512, 512, 1.0f, 1, 0, 0, 0);
    gemm_bf16<<<dim3(16, 157), 32, 0, stream>>>(
        summB, skwT, sk_b, skB, 2500, 512, 512, 1.0f, 1, 0, 0, 0);
    gemm_bf16<<<dim3(16, 32), 32, 0, stream>>>(
        qryB, qwT, nullptr, qbuf, 512, 512, 512, 0.125f, 0, 0, 0, 0);

    // 4. logits = sq @ sk^T * HID^-0.5 (batched over b, z = 4)
    gemm_bf16<<<dim3(20, 8, 4), 32, 0, stream>>>(
        sqB, skB, nullptr, lgt, 128, 625, 512, 0.04419417382415922f, 0,
        (size_t)128 * 512, (size_t)625 * 512, (size_t)128 * 625);

    // 5. top-8 + softmax weights
    topk_kernel<<<512, 128, 0, stream>>>(lgt, cany, tki, wts);

    // 6. fused KV-projection + attention (dominant 137 GFLOP of WMMA)
    attn_kernel<<<dim3(8, 128, 4), 256, 101376, stream>>>(
        memories, mask, kvwT, qbuf, posT, tki, obufB);

    // 7. output projection
    gemm_bf16<<<dim3(16, 256), 32, 0, stream>>>(
        obufB, outwT, out_b, opro, 4096, 512, 512, 1.0f, 0, 0, 0, 0);

    // 8. weighted sum over k + fc2
    final_kernel<<<512, 256, 0, stream>>>(opro, wts, fc2_w, fc2_b, out);
}
